// GAT_48404281426489
// MI455X (gfx1250) — matrix operand order repaired and checked by reading, not compiled, not run on hardware
//
#include <hip/hip_runtime.h>
#include <hip/hip_fp16.h>

typedef __attribute__((ext_vector_type(16))) _Float16 v16h;
typedef __attribute__((ext_vector_type(8)))  _Float16 v8h;
typedef __attribute__((ext_vector_type(8)))  float    v8f;

#define N_NODES   50000
#define N_EDGES   800000
#define E_TOT     850000       // edges + self loops
#define HEADS     4
#define DIM_EMB   64
#define HC        256          // HEADS*DIM_EMB
#define DCAT      768          // 3 layers * HC
#define NGRAPH    500
#define NEG_SLOPE 0.2f

// ---------------------------------------------------------------------------
// Generic fill
// ---------------------------------------------------------------------------
__global__ void fill_kernel(float* __restrict__ p, long n, float v) {
    long i = (long)blockIdx.x * blockDim.x + threadIdx.x;
    if (i < n) p[i] = v;
}

// Zero a [N_NODES, HC] slice embedded in hcat (row stride DCAT)
__global__ void zero_slice_kernel(float* __restrict__ base) {
    long i = (long)blockIdx.x * blockDim.x + threadIdx.x;
    if (i >= (long)N_NODES * HC) return;
    int n = (int)(i >> 8);
    int c = (int)(i & 255);
    base[(size_t)n * DCAT + c] = 0.0f;
}

// out[n, c] += b[c] on a slice with row stride DCAT
__global__ void bias_add_kernel(float* __restrict__ base, const float* __restrict__ b) {
    long i = (long)blockIdx.x * blockDim.x + threadIdx.x;
    if (i >= (long)N_NODES * HC) return;
    int n = (int)(i >> 8);
    int c = (int)(i & 255);
    base[(size_t)n * DCAT + c] += b[c];
}

// ---------------------------------------------------------------------------
// WMMA GEMM: C[M x N] = A[M x K] * B[K x N] (+bias)(+relu)
// Row-major fp32 in memory, f16 WMMA with f32 accumulate.
// Block = 256 threads = 8 waves; tile = 128 rows x 64 cols; K step 32.
// grid.x = ceil(M/128), grid.y = N/64.  K % 32 == 0, N % 64 == 0 required.
// ---------------------------------------------------------------------------
__global__ __launch_bounds__(256) void gemm_wmma_kernel(
    const float* __restrict__ A, int lda,
    const float* __restrict__ B, int ldb,
    const float* __restrict__ bias,
    float* __restrict__ C, int ldc,
    int M, int K, int relu)
{
    __shared__ __align__(32) _Float16 ldsB[64 * 32];  // B tile, transposed: [col][k]

    const int tid    = threadIdx.x;
    const int lane   = tid & 31;
    const int wave   = tid >> 5;
    const int tile_m = blockIdx.x * 128;
    const int tile_n = blockIdx.y * 64;

    const int  arow_m = tile_m + wave * 16 + (lane & 15);   // A-fragment row for this lane
    const int  klo    = (lane & 16) ? 16 : 0;               // A K sub-offset
    const int  kb     = (lane & 16) ? 16 : 0;               // B K sub-offset
    const int  colf   = lane & 15;
    const bool arow_ok = (arow_m < M);

    // staging ownership: thread -> (one col, 8-K strip)
    const int scol = tid & 63;          // 0..63
    const int sk0  = (tid >> 6) * 8;    // 0, 8, 16, 24

    v8f acc0 = {}, acc1 = {}, acc2 = {}, acc3 = {};

    for (int kk = 0; kk < K; kk += 32) {
        // --- stage B[kk..kk+32) x [tile_n..tile_n+64) transposed into LDS ---
        // 8 independent strided loads (clause) -> 1x ds_store_b128
        {
            const float* bcol = B + (size_t)(kk + sk0) * ldb + tile_n + scol;
            v8h pk;
#pragma unroll
            for (int k = 0; k < 8; ++k)
                pk[k] = (_Float16)bcol[(size_t)k * ldb];
            *(v8h*)(ldsB + scol * 32 + sk0) = pk;   // byte off = scol*64 + sk0*2 : 16B aligned
        }
        __syncthreads();

        // --- A fragment: one contiguous 16-float run per lane (k = klo + s) ---
        v16h a = {};
        if (arow_ok) {
            const float* ar = A + (size_t)arow_m * lda + kk + klo;
            __builtin_prefetch(ar + 32, 0, 1);   // next K-tile (global_prefetch_b8)
            float4 p0 = *(const float4*)(ar);
            float4 p1 = *(const float4*)(ar + 4);
            float4 p2 = *(const float4*)(ar + 8);
            float4 p3 = *(const float4*)(ar + 12);
            a[0]  = (_Float16)p0.x; a[1]  = (_Float16)p0.y;
            a[2]  = (_Float16)p0.z; a[3]  = (_Float16)p0.w;
            a[4]  = (_Float16)p1.x; a[5]  = (_Float16)p1.y;
            a[6]  = (_Float16)p1.z; a[7]  = (_Float16)p1.w;
            a[8]  = (_Float16)p2.x; a[9]  = (_Float16)p2.y;
            a[10] = (_Float16)p2.z; a[11] = (_Float16)p2.w;
            a[12] = (_Float16)p3.x; a[13] = (_Float16)p3.y;
            a[14] = (_Float16)p3.z; a[15] = (_Float16)p3.w;
        }

        // --- B fragments from LDS: 16 contiguous halfs per lane per subtile ---
        const _Float16* bp = ldsB + kb;
        v16h b0 = *(const v16h*)(bp + (colf +  0) * 32);
        v16h b1 = *(const v16h*)(bp + (colf + 16) * 32);
        v16h b2 = *(const v16h*)(bp + (colf + 32) * 32);
        v16h b3 = *(const v16h*)(bp + (colf + 48) * 32);

        acc0 = __builtin_amdgcn_wmma_f32_16x16x32_f16(false, a, false, b0, (short)0, acc0, false, false);
        acc1 = __builtin_amdgcn_wmma_f32_16x16x32_f16(false, a, false, b1, (short)0, acc1, false, false);
        acc2 = __builtin_amdgcn_wmma_f32_16x16x32_f16(false, a, false, b2, (short)0, acc2, false, false);
        acc3 = __builtin_amdgcn_wmma_f32_16x16x32_f16(false, a, false, b3, (short)0, acc3, false, false);
        __syncthreads();
    }

    // --- store C (+bias, relu). D layout: col = lane&15, row = i + 8*(lane>=16) ---
    const int colb  = tile_n + colf;
    const int rbase = tile_m + wave * 16 + ((lane & 16) ? 8 : 0);
    float bv0 = 0.f, bv1 = 0.f, bv2 = 0.f, bv3 = 0.f;
    if (bias) {
        bv0 = bias[colb +  0]; bv1 = bias[colb + 16];
        bv2 = bias[colb + 32]; bv3 = bias[colb + 48];
    }
#pragma unroll
    for (int i = 0; i < 8; ++i) {
        int r = rbase + i;
        if (r < M) {
            float v0 = acc0[i] + bv0;
            float v1 = acc1[i] + bv1;
            float v2 = acc2[i] + bv2;
            float v3 = acc3[i] + bv3;
            if (relu) {
                v0 = fmaxf(v0, 0.f); v1 = fmaxf(v1, 0.f);
                v2 = fmaxf(v2, 0.f); v3 = fmaxf(v3, 0.f);
            }
            float* crow = C + (size_t)r * ldc + colb;
            crow[0]  = v0; crow[16] = v1;
            crow[32] = v2; crow[48] = v3;
        }
    }
}

// ---------------------------------------------------------------------------
// asrc[n,h] = sum_c hp[n,h,c]*a_src[h,c] ; adst likewise
// ---------------------------------------------------------------------------
__global__ void attn_coef_kernel(const float* __restrict__ hp,
                                 const float* __restrict__ a_src,
                                 const float* __restrict__ a_dst,
                                 float* __restrict__ asrc,
                                 float* __restrict__ adst)
{
    int i = blockIdx.x * blockDim.x + threadIdx.x;   // n*HEADS + h
    if (i >= N_NODES * HEADS) return;
    int n = i >> 2, h = i & 3;
    const float* v  = hp    + (size_t)n * HC + h * DIM_EMB;
    const float* as = a_src + h * DIM_EMB;
    const float* ad = a_dst + h * DIM_EMB;
    float s1 = 0.f, s2 = 0.f;
#pragma unroll 4
    for (int c = 0; c < DIM_EMB; c += 4) {
        float4 x = *(const float4*)(v + c);
        float4 p = *(const float4*)(as + c);
        float4 q = *(const float4*)(ad + c);
        s1 += x.x * p.x + x.y * p.y + x.z * p.z + x.w * p.w;
        s2 += x.x * q.x + x.y * q.y + x.z * q.z + x.w * q.w;
    }
    asrc[i] = s1;
    adst[i] = s2;
}

// ---------------------------------------------------------------------------
// Order-preserving float atomic max (int punning; init with -inf)
// ---------------------------------------------------------------------------
__device__ inline void atomicMaxF(float* addr, float v) {
    int iv = __float_as_int(v);
    if (iv >= 0) atomicMax((int*)addr, iv);
    else         atomicMin((unsigned int*)addr, (unsigned int)iv);
}

__device__ inline void edge_endpoints(int e, const int* __restrict__ ei, int& s, int& d) {
    if (e < N_EDGES) { s = ei[e]; d = ei[N_EDGES + e]; }
    else             { s = d = e - N_EDGES; }          // self loop
}

// pass 1: segment max of leaky_relu(asrc[s]+adst[d]) over dst
__global__ void edge_max_kernel(const int* __restrict__ ei,
                                const float* __restrict__ asrc,
                                const float* __restrict__ adst,
                                float* __restrict__ emax)
{
    int e = blockIdx.x * blockDim.x + threadIdx.x;
    if (e >= E_TOT) return;
    int s, d; edge_endpoints(e, ei, s, d);
    float4 av = *(const float4*)(asrc + (size_t)s * 4);
    float4 dv = *(const float4*)(adst + (size_t)d * 4);
    float ev[4] = {av.x + dv.x, av.y + dv.y, av.z + dv.z, av.w + dv.w};
#pragma unroll
    for (int h = 0; h < 4; ++h) {
        float x = ev[h];
        x = (x > 0.f) ? x : NEG_SLOPE * x;
        atomicMaxF(&emax[(size_t)d * 4 + h], x);
    }
}

// pass 2: ee = exp(e - emax[dst]); denom[dst] += ee
__global__ void edge_expsum_kernel(const int* __restrict__ ei,
                                   const float* __restrict__ asrc,
                                   const float* __restrict__ adst,
                                   const float* __restrict__ emax,
                                   float* __restrict__ ee,
                                   float* __restrict__ denom)
{
    int e = blockIdx.x * blockDim.x + threadIdx.x;
    if (e >= E_TOT) return;
    int s, d; edge_endpoints(e, ei, s, d);
    float4 av = *(const float4*)(asrc + (size_t)s * 4);
    float4 dv = *(const float4*)(adst + (size_t)d * 4);
    float4 mv = *(const float4*)(emax + (size_t)d * 4);
    float ev[4] = {av.x + dv.x, av.y + dv.y, av.z + dv.z, av.w + dv.w};
    float mm[4] = {mv.x, mv.y, mv.z, mv.w};
#pragma unroll
    for (int h = 0; h < 4; ++h) {
        float x = ev[h];
        x = (x > 0.f) ? x : NEG_SLOPE * x;
        float xe = __expf(x - mm[h]);
        ee[(size_t)e * 4 + h] = xe;
        atomicAdd(&denom[(size_t)d * 4 + h], xe);
    }
}

// pass 3: out[dst, h, :] += alpha * hp[src, h, :]   (out has row stride DCAT)
__global__ void edge_aggr_kernel(const int* __restrict__ ei,
                                 const float* __restrict__ ee,
                                 const float* __restrict__ denom,
                                 const float* __restrict__ hp,
                                 float* __restrict__ out)
{
    long i = (long)blockIdx.x * blockDim.x + threadIdx.x;   // e*HEADS + h
    if (i >= (long)E_TOT * HEADS) return;
    int e = (int)(i >> 2), h = (int)(i & 3);
    int s, d; edge_endpoints(e, ei, s, d);
    float alpha = ee[(size_t)e * 4 + h] / (denom[(size_t)d * 4 + h] + 1e-16f);
    const float* v = hp  + (size_t)s * HC   + h * DIM_EMB;
    float*       o = out + (size_t)d * DCAT + h * DIM_EMB;
#pragma unroll 4
    for (int c = 0; c < DIM_EMB; c += 4) {
        float4 x = *(const float4*)(v + c);
        atomicAdd(o + c + 0, alpha * x.x);
        atomicAdd(o + c + 1, alpha * x.y);
        atomicAdd(o + c + 2, alpha * x.z);
        atomicAdd(o + c + 3, alpha * x.w);
    }
}

// ---------------------------------------------------------------------------
// pooled[batch[n], c] = max over nodes (init -inf)
// ---------------------------------------------------------------------------
__global__ void pool_max_kernel(const float* __restrict__ hcat,
                                const int* __restrict__ batch,
                                float* __restrict__ pooled)
{
    long i = (long)blockIdx.x * blockDim.x + threadIdx.x;
    if (i >= (long)N_NODES * DCAT) return;
    int n = (int)(i / DCAT);
    int c = (int)(i % DCAT);
    atomicMaxF(&pooled[(size_t)batch[n] * DCAT + c], hcat[i]);
}

// final layer: out[g] = z2[g,:] . W3 + b3
__global__ void fc3_kernel(const float* __restrict__ z2,
                           const float* __restrict__ W3,
                           const float* __restrict__ b3,
                           float* __restrict__ out)
{
    int g = blockIdx.x * blockDim.x + threadIdx.x;
    if (g >= NGRAPH) return;
    float s = b3[0];
#pragma unroll 8
    for (int c = 0; c < DIM_EMB; ++c) s += z2[(size_t)g * DIM_EMB + c] * W3[c];
    out[g] = s;
}

// ---------------------------------------------------------------------------
// Host launch
// ---------------------------------------------------------------------------
extern "C" void kernel_launch(void* const* d_in, const int* in_sizes, int n_in,
                              void* d_out, int out_size, void* d_ws, size_t ws_size,
                              hipStream_t stream) {
    (void)in_sizes; (void)n_in; (void)out_size; (void)ws_size;

    const float* x     = (const float*)d_in[0];
    const int*   ei    = (const int*)d_in[1];
    const int*   batch = (const int*)d_in[2];
    const float* Wl[3] = {(const float*)d_in[3],  (const float*)d_in[7],  (const float*)d_in[11]};
    const float* As[3] = {(const float*)d_in[4],  (const float*)d_in[8],  (const float*)d_in[12]};
    const float* Ad[3] = {(const float*)d_in[5],  (const float*)d_in[9],  (const float*)d_in[13]};
    const float* Bb[3] = {(const float*)d_in[6],  (const float*)d_in[10], (const float*)d_in[14]};
    const float* W1 = (const float*)d_in[15]; const float* b1 = (const float*)d_in[16];
    const float* W2 = (const float*)d_in[17]; const float* b2 = (const float*)d_in[18];
    const float* W3 = (const float*)d_in[19]; const float* b3 = (const float*)d_in[20];
    float* out = (float*)d_out;

    // workspace layout (floats)
    float* ws = (float*)d_ws;
    size_t o = 0;
    float* hcat   = ws + o; o += (size_t)N_NODES * DCAT;   // 3-layer concat
    float* hp     = ws + o; o += (size_t)N_NODES * HC;     // projected features
    float* asrc   = ws + o; o += (size_t)N_NODES * HEADS;
    float* adst   = ws + o; o += (size_t)N_NODES * HEADS;
    float* emax   = ws + o; o += (size_t)N_NODES * HEADS;
    float* denom  = ws + o; o += (size_t)N_NODES * HEADS;
    float* eebuf  = ws + o; o += (size_t)E_TOT   * HEADS;
    float* pooled = ws + o; o += (size_t)NGRAPH  * DCAT;
    float* z1     = ws + o; o += (size_t)NGRAPH  * HC;
    float* z2     = ws + o; o += (size_t)NGRAPH  * DIM_EMB;

    const float NEG_INF = -__builtin_huge_valf();

    const dim3 gemm_blk(256);
    const dim3 gemm_grid_nodes((N_NODES + 127) / 128, HC / 64);      // (391, 4)
    const int  THR = 256;
    const int  g_nh    = (N_NODES * HEADS + THR - 1) / THR;
    const int  g_edge  = (E_TOT + THR - 1) / THR;
    const int  g_eh    = (int)(((long)E_TOT * HEADS + THR - 1) / THR);
    const int  g_slice = (int)(((long)N_NODES * HC + THR - 1) / THR);
    const int  g_pfill = (int)(((long)NGRAPH * DCAT + THR - 1) / THR);
    const int  g_pool  = (int)(((long)N_NODES * DCAT + THR - 1) / THR);

    for (int l = 0; l < 3; ++l) {
        const float* Ain = (l == 0) ? x : (hcat + (size_t)(l - 1) * HC);
        const int lda = (l == 0) ? 128 : DCAT;
        const int K   = (l == 0) ? 128 : HC;
        float* oslice = hcat + (size_t)l * HC;

        // hp = Ain @ W[l]   (bias applied after aggregation, as in reference)
        gemm_wmma_kernel<<<gemm_grid_nodes, gemm_blk, 0, stream>>>(
            Ain, lda, Wl[l], HC, nullptr, hp, HC, N_NODES, K, 0);

        attn_coef_kernel<<<g_nh, THR, 0, stream>>>(hp, As[l], Ad[l], asrc, adst);

        fill_kernel<<<g_nh, THR, 0, stream>>>(emax,  (long)N_NODES * HEADS, NEG_INF);
        fill_kernel<<<g_nh, THR, 0, stream>>>(denom, (long)N_NODES * HEADS, 0.0f);
        zero_slice_kernel<<<g_slice, THR, 0, stream>>>(oslice);

        edge_max_kernel<<<g_edge, THR, 0, stream>>>(ei, asrc, adst, emax);
        edge_expsum_kernel<<<g_edge, THR, 0, stream>>>(ei, asrc, adst, emax, eebuf, denom);
        edge_aggr_kernel<<<g_eh, THR, 0, stream>>>(ei, eebuf, denom, hp, oslice);
        bias_add_kernel<<<g_slice, THR, 0, stream>>>(oslice, Bb[l]);
    }

    // graph pooling: segment max over batch
    fill_kernel<<<g_pfill, THR, 0, stream>>>(pooled, (long)NGRAPH * DCAT, NEG_INF);
    pool_max_kernel<<<g_pool, THR, 0, stream>>>(hcat, batch, pooled);

    // MLP head: z1 = relu(pooled@W1+b1), z2 = relu(z1@W2+b2), out = z2@W3+b3
    gemm_wmma_kernel<<<dim3((NGRAPH + 127) / 128, HC / 64), gemm_blk, 0, stream>>>(
        pooled, DCAT, W1, HC, b1, z1, HC, NGRAPH, DCAT, 1);
    gemm_wmma_kernel<<<dim3((NGRAPH + 127) / 128, DIM_EMB / 64), gemm_blk, 0, stream>>>(
        z1, HC, W2, DIM_EMB, b2, z2, DIM_EMB, NGRAPH, HC, 1);
    fc3_kernel<<<(NGRAPH + THR - 1) / THR, THR, 0, stream>>>(z2, W3, b3, out);
}
